// WorldGraphEncoder_17875653886604
// MI455X (gfx1250) — matrix operand order repaired and checked by reading, not compiled, not run on hardware
//
#include <hip/hip_runtime.h>
#include <hip/hip_bf16.h>

// ---------------------------------------------------------------------------
// WorldGraphEncoder for MI455X (gfx1250): f16 WMMA (16x16x32) everywhere,
// fused 2-layer MLPs with LDS hidden, TDM (tensor_load_to_lds) weight staging,
// fused gate*msg scatter via f32 atomics. Workspace requirement: ~73 MB.
// ---------------------------------------------------------------------------

typedef __attribute__((ext_vector_type(16))) _Float16 v16h;
typedef __attribute__((ext_vector_type(8)))  _Float16 v8h;
typedef __attribute__((ext_vector_type(8)))  float    v8f;
typedef unsigned int u32x4 __attribute__((ext_vector_type(4)));
typedef int          i32x4 __attribute__((ext_vector_type(4)));
typedef int          i32x8 __attribute__((ext_vector_type(8)));

__device__ __forceinline__ float gelu_exact(float x) {
    return 0.5f * x * (1.0f + erff(x * 0.7071067811865476f));
}
__device__ __forceinline__ float sigmoidf_(float x) {
    return 1.0f / (1.0f + __expf(-x));
}
__device__ __forceinline__ v16h cat16(v8h lo, v8h hi) {
    v16h r;
#pragma unroll
    for (int i = 0; i < 8; i++) { r[i] = lo[i]; r[i + 8] = hi[i]; }
    return r;
}
__device__ __forceinline__ v8f zero8() {
    v8f r;
#pragma unroll
    for (int i = 0; i < 8; i++) r[i] = 0.0f;
    return r;
}
__device__ __forceinline__ v8h zero8h() {
    v8h r;
#pragma unroll
    for (int i = 0; i < 8; i++) r[i] = (_Float16)0;
    return r;
}
// A fragment 16x32 f16: lane<16 -> row=lane, K in {0..15};
//                       lane>=16 -> row=lane-16, K in {16..31}
__device__ __forceinline__ v16h load_a(const _Float16* p, int khalf) {
    v8h lo = *(const v8h*)(p + 2 * khalf);
    v8h hi = *(const v8h*)(p + 2 * khalf + 8);
    return cat16(lo, hi);
}
// B fragment 32x16 f16 from Wt[n][K]: lane: n=lane&15, k = kb*32 +
// (lane>>4)*16 .. +15 contiguous. p fully precomputed.
__device__ __forceinline__ v16h load_b(const _Float16* p) {
    v8h lo = *(const v8h*)(p);
    v8h hi = *(const v8h*)(p + 8);
    return cat16(lo, hi);
}
__device__ __forceinline__ v8f wmma32(v16h a, v16h b, v8f c) {
    return __builtin_amdgcn_wmma_f32_16x16x32_f16(false, a, false, b,
                                                  (short)0, c, false, false);
}

// ---------------------------------------------------------------------------
// TDM: one-shot DMA of a contiguous block (multiple of 8B, <=512KB) from
// global to LDS. D# per CDNA5 ISA ch.8: 1-row tile of n8 8-byte elements.
// Issue from one wave; caller must s_wait_tensorcnt + __syncthreads().
// ---------------------------------------------------------------------------
__device__ __forceinline__ void tdm_load_to_lds(const void* gptr, void* lptr,
                                                unsigned bytes) {
    unsigned long long ga = (unsigned long long)gptr;
    unsigned lds = (unsigned)(unsigned long long)lptr;  // flat low 32b = LDS off
    unsigned n8 = bytes >> 3;
    u32x4 g0;
    g0[0] = 1u;                                   // count=1 (valid user D#)
    g0[1] = lds;                                  // lds_addr
    g0[2] = (unsigned)(ga & 0xffffffffu);         // global_addr[31:0]
    g0[3] = (unsigned)((ga >> 32) & 0x01ffffffu)  // global_addr[56:32]
            | (2u << 30);                         // type = 2 ("image")
    i32x8 g1;
    g1[0] = (int)(3u << 16);                      // data_size = 8 bytes
    g1[1] = (int)((n8 & 0xffffu) << 16);          // tensor_dim0[15:0]
    g1[2] = (int)((n8 >> 16) & 0xffffu)           // tensor_dim0[31:16]
            | (int)(1u << 16);                    // tensor_dim1 = 1
    g1[3] = (int)((n8 & 0xffffu) << 16);          // tile_dim0 = n8
    g1[4] = 1;                                    // tile_dim1 = 1
    g1[5] = (int)n8;                              // tensor_dim0_stride = n8
    g1[6] = 0;
    g1[7] = 0;
    i32x4 z4;
    z4[0] = 0; z4[1] = 0; z4[2] = 0; z4[3] = 0;
#if defined(__clang_major__) && (__clang_major__ >= 23)
    i32x8 z8;
#pragma unroll
    for (int i = 0; i < 8; i++) z8[i] = 0;
    __builtin_amdgcn_tensor_load_to_lds(g0, g1, z4, z4, z8, 0);
#else
    __builtin_amdgcn_tensor_load_to_lds(g0, g1, z4, z4, 0);
#endif
}

// ---------------------------------------------------------------------------
// Prep: transpose f32 [K,Nc] -> f16 [Nc,K]
// ---------------------------------------------------------------------------
__global__ void wt_prep_kernel(const float* __restrict__ src,
                               _Float16* __restrict__ dst, int K, int Nc) {
    int i = blockIdx.x * blockDim.x + threadIdx.x;
    int total = K * Nc;
    if (i >= total) return;
    int n = i / K, k = i - n * K;
    dst[i] = (_Float16)src[k * Nc + n];
}
__global__ void cvt_f16_kernel(const float* __restrict__ src,
                               _Float16* __restrict__ dst, int total) {
    int i = blockIdx.x * blockDim.x + threadIdx.x;
    if (i < total) dst[i] = (_Float16)src[i];
}

// ---------------------------------------------------------------------------
// Byte signature pooling: pooled16[row][0:128]=mean, [128:256]=max, [256:384]=first
// ---------------------------------------------------------------------------
__global__ __launch_bounds__(128) void byte_pool_kernel(
    const int* __restrict__ sig_bytes, const float* __restrict__ byte_emb,
    const float* __restrict__ pos_emb, _Float16* __restrict__ pooled, int rows) {
    int row = blockIdx.x;
    if (row >= rows) return;
    int d = threadIdx.x;
    const int* sb = sig_bytes + (size_t)row * 64;
    float s = 0.0f, mx = -3.4e38f, first = 0.0f;
#pragma unroll 4
    for (int l = 0; l < 64; l++) {
        int b = sb[l];
        float v = byte_emb[b * 128 + d] + pos_emb[l * 128 + d];
        s += v;
        mx = fmaxf(mx, v);
        if (l == 0) first = v;
    }
    _Float16* o = pooled + (size_t)row * 384;
    o[d]       = (_Float16)(s * (1.0f / 64.0f));
    o[128 + d] = (_Float16)mx;
    o[256 + d] = (_Float16)first;
}

// ---------------------------------------------------------------------------
// Fused 2-layer MLP, K1=384 -> GELU -> K2=128.
// MODE 0: +LayerNorm epilogue, f16 output.  MODE 1: plain f32 output.
// One 16x128 output tile per wave; hidden roundtrips via per-wave LDS scratch.
// ---------------------------------------------------------------------------
template <int MODE>
__global__ __launch_bounds__(256) void mlp384_kernel(
    const _Float16* __restrict__ A,    // [rows][384] f16
    const _Float16* __restrict__ W1t,  // [128][384] f16
    const _Float16* __restrict__ W2t,  // [128][128] f16
    const float* __restrict__ b1, const float* __restrict__ b2,
    const float* __restrict__ lng, const float* __restrict__ lnb,
    _Float16* __restrict__ out16, float* __restrict__ out32, int tiles) {
    __shared__ _Float16 sW1t[128 * 384];
    __shared__ _Float16 sScr[8 * 16 * 128];
    if (threadIdx.x < 32) {  // wave 0 drives the TDM, then waits on TENSORcnt
        tdm_load_to_lds(W1t, sW1t, 128 * 384 * 2);
        __builtin_amdgcn_s_wait_tensorcnt(0);
    }
    __syncthreads();
    const int lane = threadIdx.x & 31;
    const int wave = threadIdx.x >> 5;
    const int n_ = lane & 15;
    const int hsel = lane >> 4;
    const int khA = hsel * 8;
    const int khB = hsel * 16;
    _Float16* scr = sScr + wave * (16 * 128);
    int gw = blockIdx.x * 8 + wave;
    int gstride = gridDim.x * 8;
    for (int t = gw; t < tiles; t += gstride) {
        const _Float16* arow = A + (size_t)(t * 16 + n_) * 384;
        v8f acc[8];
#pragma unroll
        for (int nt = 0; nt < 8; nt++) acc[nt] = zero8();
#pragma unroll
        for (int kb = 0; kb < 12; kb++) {
            v16h a = load_a(arow + kb * 32, khA);
#pragma unroll
            for (int nt = 0; nt < 8; nt++) {
                v16h b = load_b(sW1t + (nt * 16 + n_) * 384 + kb * 32 + khB);
                acc[nt] = wmma32(a, b, acc[nt]);
            }
        }
        // layer-1 epilogue: bias + GELU -> LDS scratch (f16), D->A transpose
#pragma unroll
        for (int nt = 0; nt < 8; nt++) {
            int col = nt * 16 + n_;
            float bb = b1[col];
#pragma unroll
            for (int v = 0; v < 8; v++) {
                int m = v + hsel * 8;
                scr[m * 128 + col] = (_Float16)gelu_exact(acc[nt][v] + bb);
            }
        }
        // layer 2: K=128
        v8f acc2[8];
#pragma unroll
        for (int nt = 0; nt < 8; nt++) acc2[nt] = zero8();
        const _Float16* arow2 = scr + n_ * 128;
#pragma unroll
        for (int kb = 0; kb < 4; kb++) {
            v16h a = load_a(arow2 + kb * 32, khA);
#pragma unroll
            for (int nt = 0; nt < 8; nt++) {
                v16h b = load_b(W2t + (nt * 16 + n_) * 128 + kb * 32 + khB);
                acc2[nt] = wmma32(a, b, acc2[nt]);
            }
        }
        if (MODE == 0) {
            // +b2 then LayerNorm over the 128-wide row held across 16 lanes
            float rs[8];
#pragma unroll
            for (int v = 0; v < 8; v++) rs[v] = 0.0f;
#pragma unroll
            for (int nt = 0; nt < 8; nt++) {
                float bb = b2[nt * 16 + n_];
#pragma unroll
                for (int v = 0; v < 8; v++) { acc2[nt][v] += bb; rs[v] += acc2[nt][v]; }
            }
#pragma unroll
            for (int v = 0; v < 8; v++) {
                float s = rs[v];
                s += __shfl_xor(s, 1); s += __shfl_xor(s, 2);
                s += __shfl_xor(s, 4); s += __shfl_xor(s, 8);
                rs[v] = s * (1.0f / 128.0f);
            }
            float rv[8];
#pragma unroll
            for (int v = 0; v < 8; v++) rv[v] = 0.0f;
#pragma unroll
            for (int nt = 0; nt < 8; nt++)
#pragma unroll
                for (int v = 0; v < 8; v++) {
                    float dlt = acc2[nt][v] - rs[v];
                    rv[v] += dlt * dlt;
                }
#pragma unroll
            for (int v = 0; v < 8; v++) {
                float q = rv[v];
                q += __shfl_xor(q, 1); q += __shfl_xor(q, 2);
                q += __shfl_xor(q, 4); q += __shfl_xor(q, 8);
                rv[v] = rsqrtf(q * (1.0f / 128.0f) + 1e-5f);
            }
#pragma unroll
            for (int nt = 0; nt < 8; nt++) {
                int col = nt * 16 + n_;
                float gg = lng[col], bb = lnb[col];
#pragma unroll
                for (int v = 0; v < 8; v++) {
                    int m = v + hsel * 8;
                    out16[(size_t)(t * 16 + m) * 128 + col] =
                        (_Float16)((acc2[nt][v] - rs[v]) * rv[v] * gg + bb);
                }
            }
        } else {
#pragma unroll
            for (int nt = 0; nt < 8; nt++) {
                int col = nt * 16 + n_;
                float bb = b2[col];
#pragma unroll
                for (int v = 0; v < 8; v++) {
                    int m = v + hsel * 8;
                    out32[(size_t)(t * 16 + m) * 128 + col] = acc2[nt][v] + bb;
                }
            }
        }
    }
}

// ---------------------------------------------------------------------------
// Block-level LayerNorm helper over 128 threads (one element each)
// ---------------------------------------------------------------------------
__device__ __forceinline__ float block_ln_128(float x, const float* g,
                                              const float* b, int d) {
    __shared__ float red[4];
    int lane = threadIdx.x & 31, wave = threadIdx.x >> 5;
    float s = x;
#pragma unroll
    for (int m = 16; m >= 1; m >>= 1) s += __shfl_xor(s, m);
    if (lane == 0) red[wave] = s;
    __syncthreads();
    float mean = (red[0] + red[1] + red[2] + red[3]) * (1.0f / 128.0f);
    __syncthreads();
    float dlt = x - mean;
    float q = dlt * dlt;
#pragma unroll
    for (int m = 16; m >= 1; m >>= 1) q += __shfl_xor(q, m);
    if (lane == 0) red[wave] = q;
    __syncthreads();
    float var = (red[0] + red[1] + red[2] + red[3]) * (1.0f / 128.0f);
    return dlt * rsqrtf(var + 1e-5f) * g[d] + b[d];
}

// h = LN(signature + 0.25*(pred + arity + mean(term[args]))) ; also f16 mirror
__global__ __launch_bounds__(128) void bucket_ln_kernel(
    const int* __restrict__ pred_idx, const int* __restrict__ arity_idx,
    const int* __restrict__ arg_idx, const float* __restrict__ pred_emb,
    const float* __restrict__ arity_emb, const float* __restrict__ term_emb,
    const float* __restrict__ an_g, const float* __restrict__ an_b,
    float* __restrict__ h32, _Float16* __restrict__ h16, int Nn) {
    int n = blockIdx.x;
    if (n >= Nn) return;
    int d = threadIdx.x;
    float sig = h32[(size_t)n * 128 + d];
    float bk = pred_emb[(size_t)pred_idx[n] * 128 + d] +
               arity_emb[(size_t)arity_idx[n] * 128 + d];
    float ts = 0.0f;
    const int* ar = arg_idx + (size_t)n * 8;
#pragma unroll
    for (int a = 0; a < 8; a++) ts += term_emb[(size_t)ar[a] * 128 + d];
    float x = sig + 0.25f * (bk + ts * 0.125f);
    float y = block_ln_128(x, an_g, an_b, d);
    h32[(size_t)n * 128 + d] = y;
    h16[(size_t)n * 128 + d] = (_Float16)y;
}

// h = LN(h + agg) ; refresh f16 mirror
__global__ __launch_bounds__(128) void ln_layer_kernel(
    float* __restrict__ h32, const float* __restrict__ agg,
    const float* __restrict__ g, const float* __restrict__ b,
    _Float16* __restrict__ h16, int Nn) {
    int n = blockIdx.x;
    if (n >= Nn) return;
    int d = threadIdx.x;
    float x = h32[(size_t)n * 128 + d] + agg[(size_t)n * 128 + d];
    float y = block_ln_128(x, g, b, d);
    h32[(size_t)n * 128 + d] = y;
    h16[(size_t)n * 128 + d] = (_Float16)y;
}

// ---------------------------------------------------------------------------
// Fused edge kernel: per 16-edge tile, gate MLP (384->128->1, sigmoid) and
// msg MLP (256->128->128), scatter gate*msg into agg with f32 atomics.
// msg_W1t staged in LDS via TDM; gate_W1t / msg_W2t served from L2 (hot).
// ---------------------------------------------------------------------------
__global__ __launch_bounds__(256) void edge_kernel(
    const _Float16* __restrict__ h16, const _Float16* __restrict__ rel16,
    const int* __restrict__ src_idx, const int* __restrict__ dst_idx,
    const int* __restrict__ rel_idx, const _Float16* __restrict__ msgW1t,
    const _Float16* __restrict__ msgW2t, const _Float16* __restrict__ gateW1t,
    const float* __restrict__ msg_b1, const float* __restrict__ msg_b2,
    const float* __restrict__ gate_b1, const float* __restrict__ gate_W2,
    const float* __restrict__ gate_b2, float* __restrict__ agg, int tiles) {
    __shared__ _Float16 sW1t[128 * 256];
    __shared__ _Float16 sScr[8 * 16 * 128];
    __shared__ float sGb1[128], sMb1[128], sMb2[128];
    __shared__ _Float16 sGw2h[128];
    if (threadIdx.x < 32) {  // wave 0 drives the TDM, then waits on TENSORcnt
        tdm_load_to_lds(msgW1t, sW1t, 128 * 256 * 2);
        __builtin_amdgcn_s_wait_tensorcnt(0);
    }
    for (int i = threadIdx.x; i < 128; i += blockDim.x) {
        sGb1[i] = gate_b1[i];
        sMb1[i] = msg_b1[i];
        sMb2[i] = msg_b2[i];
        sGw2h[i] = (_Float16)gate_W2[i];
    }
    __syncthreads();
    const int lane = threadIdx.x & 31;
    const int wave = threadIdx.x >> 5;
    const int n_ = lane & 15;
    const int hsel = lane >> 4;
    const int khA = hsel * 8;
    const int khB = hsel * 16;
    _Float16* scr = sScr + wave * (16 * 128);
    const float gb2 = gate_b2[0];
    int gw = blockIdx.x * 8 + wave;
    int gstride = gridDim.x * 8;
    for (int t = gw; t < tiles; t += gstride) {
        int e = t * 16 + n_;
        // prefetch index cachelines for the tile this wave handles next
        __builtin_prefetch(src_idx + e + gstride * 16, 0, 1);
        __builtin_prefetch(dst_idx + e + gstride * 16, 0, 1);
        int sidx = src_idx[e], didx = dst_idx[e], ridx = rel_idx[e];
        const _Float16* srcp = h16 + (size_t)sidx * 128;
        const _Float16* dstp = h16 + (size_t)didx * 128;
        const _Float16* relp = rel16 + (size_t)ridx * 128;
        // ---- gate layer 1: K=384 (src | dst | rel) ----
        v8f acc[8];
#pragma unroll
        for (int nt = 0; nt < 8; nt++) acc[nt] = zero8();
#pragma unroll
        for (int kb = 0; kb < 12; kb++) {
            const _Float16* ap = (kb < 4) ? (srcp + kb * 32)
                                 : (kb < 8) ? (dstp + (kb - 4) * 32)
                                            : (relp + (kb - 8) * 32);
            v16h a = load_a(ap, khA);
#pragma unroll
            for (int nt = 0; nt < 8; nt++) {
                v16h b = load_b(gateW1t + (nt * 16 + n_) * 384 + kb * 32 + khB);
                acc[nt] = wmma32(a, b, acc[nt]);
            }
        }
#pragma unroll
        for (int nt = 0; nt < 8; nt++) {
            int col = nt * 16 + n_;
            float bb = sGb1[col];
#pragma unroll
            for (int v = 0; v < 8; v++)
                scr[(v + hsel * 8) * 128 + col] = (_Float16)gelu_exact(acc[nt][v] + bb);
        }
        // gate layer 2: packed-f16 dot(hidden, w2) per edge row, sigmoid
        const v8h* hr = (const v8h*)(scr + n_ * 128 + hsel * 64);
        const v8h* wr = (const v8h*)(sGw2h + hsel * 64);
        v8h acch = zero8h();
#pragma unroll
        for (int c = 0; c < 8; c++) acch += hr[c] * wr[c];
        float dotv = 0.0f;
#pragma unroll
        for (int i = 0; i < 8; i++) dotv += (float)acch[i];
        dotv += __shfl_xor(dotv, 16);
        float gv = sigmoidf_(dotv + gb2);
        // ---- msg layer 1: K=256 (src | rel) ----
#pragma unroll
        for (int nt = 0; nt < 8; nt++) acc[nt] = zero8();
#pragma unroll
        for (int kb = 0; kb < 8; kb++) {
            const _Float16* ap = (kb < 4) ? (srcp + kb * 32) : (relp + (kb - 4) * 32);
            v16h a = load_a(ap, khA);
#pragma unroll
            for (int nt = 0; nt < 8; nt++) {
                v16h b = load_b(sW1t + (nt * 16 + n_) * 256 + kb * 32 + khB);
                acc[nt] = wmma32(a, b, acc[nt]);
            }
        }
#pragma unroll
        for (int nt = 0; nt < 8; nt++) {
            int col = nt * 16 + n_;
            float bb = sMb1[col];
#pragma unroll
            for (int v = 0; v < 8; v++)
                scr[(v + hsel * 8) * 128 + col] = (_Float16)gelu_exact(acc[nt][v] + bb);
        }
        // ---- msg layer 2: K=128 ----
        v8f acc2[8];
#pragma unroll
        for (int nt = 0; nt < 8; nt++) acc2[nt] = zero8();
        const _Float16* arow2 = scr + n_ * 128;
#pragma unroll
        for (int kb = 0; kb < 4; kb++) {
            v16h a = load_a(arow2 + kb * 32, khA);
#pragma unroll
            for (int nt = 0; nt < 8; nt++) {
                v16h b = load_b(msgW2t + (nt * 16 + n_) * 128 + kb * 32 + khB);
                acc2[nt] = wmma32(a, b, acc2[nt]);
            }
        }
        // ---- scatter: agg[dst] += gate * (msg + b2) ----
#pragma unroll
        for (int v = 0; v < 8; v++) {
            int row = v + hsel * 8;
            int de = __shfl(didx, row);
            float gr = __shfl(gv, row);
            float* aggrow = agg + (size_t)de * 128;
#pragma unroll
            for (int nt = 0; nt < 8; nt++) {
                int col = nt * 16 + n_;
                atomicAdd(aggrow + col, gr * (acc2[nt][v] + sMb2[col]));
            }
        }
    }
}

// ---------------------------------------------------------------------------
// Pool: grid partial mean/max, then single-block 256->128->128 MLP
// ---------------------------------------------------------------------------
__global__ __launch_bounds__(128) void pool_reduce_kernel(
    const float* __restrict__ h32, float* __restrict__ partS,
    float* __restrict__ partM, int Nn) {
    int d = threadIdx.x, blk = blockIdx.x;
    float s = 0.0f, m = -3.4e38f;
    for (int n = blk; n < Nn; n += gridDim.x) {
        float v = h32[(size_t)n * 128 + d];
        s += v;
        m = fmaxf(m, v);
    }
    partS[blk * 128 + d] = s;
    partM[blk * 128 + d] = m;
}

__global__ __launch_bounds__(128) void pool_final_kernel(
    const float* __restrict__ partS, const float* __restrict__ partM,
    int nparts, int Nn, const float* __restrict__ W1,
    const float* __restrict__ b1, const float* __restrict__ W2,
    const float* __restrict__ b2, float* __restrict__ outrow) {
    __shared__ float cat[256];
    __shared__ float hid[128];
    int d = threadIdx.x;
    float s = 0.0f, m = -3.4e38f;
    for (int i = 0; i < nparts; i++) {
        s += partS[i * 128 + d];
        m = fmaxf(m, partM[i * 128 + d]);
    }
    cat[d] = s / (float)Nn;
    cat[128 + d] = m;
    __syncthreads();
    float a = b1[d];
    for (int k = 0; k < 256; k++) a += cat[k] * W1[k * 128 + d];
    hid[d] = gelu_exact(a);
    __syncthreads();
    float o = b2[d];
    for (int k = 0; k < 128; k++) o += hid[k] * W2[k * 128 + d];
    outrow[d] = o;
}

// ---------------------------------------------------------------------------
extern "C" void kernel_launch(void* const* d_in, const int* in_sizes, int n_in,
                              void* d_out, int out_size, void* d_ws,
                              size_t ws_size, hipStream_t stream) {
    (void)n_in; (void)out_size; (void)ws_size;
    const int* pred_idx  = (const int*)d_in[0];
    const int* arity_idx = (const int*)d_in[1];
    const int* arg_idx   = (const int*)d_in[2];
    const int* sig_bytes = (const int*)d_in[3];
    const int* src_idx   = (const int*)d_in[4];
    const int* dst_idx   = (const int*)d_in[5];
    const int* rel_idx   = (const int*)d_in[6];
    const int P = 7;
    const float* pred_emb  = (const float*)d_in[P + 0];
    const float* term_emb  = (const float*)d_in[P + 1];
    const float* arity_emb = (const float*)d_in[P + 2];
    const float* rel_emb   = (const float*)d_in[P + 3];
    const float* byte_emb  = (const float*)d_in[P + 4];
    const float* byte_pos  = (const float*)d_in[P + 5];
    const float* txt_W1    = (const float*)d_in[P + 6];
    const float* txt_b1    = (const float*)d_in[P + 7];
    const float* txt_W2    = (const float*)d_in[P + 8];
    const float* txt_b2    = (const float*)d_in[P + 9];
    const float* atom_W1   = (const float*)d_in[P + 10];
    const float* atom_b1   = (const float*)d_in[P + 11];
    const float* atom_W2   = (const float*)d_in[P + 12];
    const float* atom_b2   = (const float*)d_in[P + 13];
    const float* sig_g     = (const float*)d_in[P + 14];
    const float* sig_b     = (const float*)d_in[P + 15];
    const float* an_g      = (const float*)d_in[P + 16];
    const float* an_b      = (const float*)d_in[P + 17];
    const float* msg_W1    = (const float*)d_in[P + 18];
    const float* msg_b1    = (const float*)d_in[P + 19];
    const float* msg_W2    = (const float*)d_in[P + 20];
    const float* msg_b2    = (const float*)d_in[P + 21];
    const float* gate_W1   = (const float*)d_in[P + 22];
    const float* gate_b1   = (const float*)d_in[P + 23];
    const float* gate_W2   = (const float*)d_in[P + 24];
    const float* gate_b2   = (const float*)d_in[P + 25];
    const float* ln_g      = (const float*)d_in[P + 26];
    const float* ln_b      = (const float*)d_in[P + 27];
    const float* pool_W1   = (const float*)d_in[P + 28];
    const float* pool_b1   = (const float*)d_in[P + 29];
    const float* pool_W2   = (const float*)d_in[P + 30];
    const float* pool_b2   = (const float*)d_in[P + 31];

    const int Nn = in_sizes[0];
    const int E  = in_sizes[4];

    // workspace carve (256B aligned)
    char* w = (char*)d_ws;
    auto carve = [&](size_t bytes) -> char* {
        char* p = w;
        w += (bytes + 255) & ~(size_t)255;
        return p;
    };
    _Float16* txtW1t    = (_Float16*)carve(384 * 128 * 2);
    _Float16* txtW2t    = (_Float16*)carve(128 * 128 * 2);
    _Float16* atomW1t   = (_Float16*)carve(384 * 128 * 2);
    _Float16* atomW2t   = (_Float16*)carve(128 * 128 * 2);
    _Float16* msgW1t[2], *msgW2t[2], *gateW1t[2];
    for (int l = 0; l < 2; l++) msgW1t[l]  = (_Float16*)carve(256 * 128 * 2);
    for (int l = 0; l < 2; l++) msgW2t[l]  = (_Float16*)carve(128 * 128 * 2);
    for (int l = 0; l < 2; l++) gateW1t[l] = (_Float16*)carve(384 * 128 * 2);
    _Float16* rel16   = (_Float16*)carve(6 * 128 * 2);
    _Float16* pooled16= (_Float16*)carve((size_t)Nn * 3 * 384 * 2);
    _Float16* sigcat16= (_Float16*)carve((size_t)Nn * 384 * 2);
    _Float16* h16     = (_Float16*)carve((size_t)Nn * 128 * 2);
    float*    h32     = (float*)   carve((size_t)Nn * 128 * 4);
    float*    agg     = (float*)   carve((size_t)Nn * 128 * 4);
    float*    partS   = (float*)   carve(256 * 128 * 4);
    float*    partM   = (float*)   carve(256 * 128 * 4);

    auto prepT = [&](const float* src, _Float16* dst, int K, int Nc) {
        int total = K * Nc;
        wt_prep_kernel<<<(total + 255) / 256, 256, 0, stream>>>(src, dst, K, Nc);
    };
    prepT(txt_W1, txtW1t, 384, 128);
    prepT(txt_W2, txtW2t, 128, 128);
    prepT(atom_W1, atomW1t, 384, 128);
    prepT(atom_W2, atomW2t, 128, 128);
    for (int l = 0; l < 2; l++) {
        prepT(msg_W1 + (size_t)l * 256 * 128, msgW1t[l], 256, 128);
        prepT(msg_W2 + (size_t)l * 128 * 128, msgW2t[l], 128, 128);
        prepT(gate_W1 + (size_t)l * 384 * 128, gateW1t[l], 384, 128);
    }
    cvt_f16_kernel<<<3, 256, 0, stream>>>(rel_emb, rel16, 6 * 128);

    // byte signature pooling -> pooled16 [N*3, 384]
    byte_pool_kernel<<<Nn * 3, 128, 0, stream>>>(sig_bytes, byte_emb, byte_pos,
                                                 pooled16, Nn * 3);
    // txt MLP + LN -> sigcat16 (= sig reshaped [N,384])
    int txtTiles = (Nn * 3) / 16;
    mlp384_kernel<0><<<384, 256, 0, stream>>>(pooled16, txtW1t, txtW2t, txt_b1,
                                              txt_b2, sig_g, sig_b, sigcat16,
                                              nullptr, txtTiles);
    // atom MLP -> h32 (signature_state, f32)
    int atomTiles = Nn / 16;
    mlp384_kernel<1><<<128, 256, 0, stream>>>(sigcat16, atomW1t, atomW2t, atom_b1,
                                              atom_b2, nullptr, nullptr, nullptr,
                                              h32, atomTiles);
    // h = LN(signature + 0.25*bucket)
    bucket_ln_kernel<<<Nn, 128, 0, stream>>>(pred_idx, arity_idx, arg_idx,
                                             pred_emb, arity_emb, term_emb, an_g,
                                             an_b, h32, h16, Nn);
    // message passing layers
    int edgeTiles = E / 16;
    for (int l = 0; l < 2; l++) {
        hipMemsetAsync(agg, 0, (size_t)Nn * 128 * 4, stream);
        edge_kernel<<<1024, 256, 0, stream>>>(
            h16, rel16, src_idx, dst_idx, rel_idx, msgW1t[l], msgW2t[l],
            gateW1t[l], msg_b1 + l * 128, msg_b2 + l * 128, gate_b1 + l * 128,
            gate_W2 + l * 128, gate_b2 + l, agg, edgeTiles);
        ln_layer_kernel<<<Nn, 128, 0, stream>>>(h32, agg, ln_g + l * 128,
                                                ln_b + l * 128, h16, Nn);
    }
    // outputs: rows 0..N-1 = h, row N = pooled MLP
    hipMemcpyAsync(d_out, h32, (size_t)Nn * 128 * 4, hipMemcpyDeviceToDevice,
                   stream);
    pool_reduce_kernel<<<256, 128, 0, stream>>>(h32, partS, partM, Nn);
    pool_final_kernel<<<1, 128, 0, stream>>>(partS, partM, 256, Nn, pool_W1,
                                             pool_b1, pool_W2, pool_b2,
                                             (float*)d_out + (size_t)Nn * 128);
}
